// Mamba2Mixer_5016521802012
// MI455X (gfx1250) — compile-verified, matrix operand order checked by reading, not run on hardware
//
#include <hip/hip_runtime.h>
#include <math.h>

#define EPS      1e-5f
#define CHUNK    256
#define NCH      8            // SEQ / CHUNK
#define BATCH    2
#define SEQ      2048
#define DMODEL   1024
#define DINNER   2048
#define DHEAD    64
#define NHEADS   32
#define DSTATE   64
#define DCONV    4
#define CONVDIM  (DINNER + 2*DSTATE)              // 2176
#define DINPROJ  (2*DINNER + 2*DSTATE + NHEADS)   // 4256
#define NPAD     4352                             // DINPROJ padded to 256-multiple
#define NTOK     (BATCH*SEQ)                      // 4096

typedef __attribute__((ext_vector_type(16))) __bf16 v16bf;
typedef __attribute__((ext_vector_type(8)))  float  v8f;

// ---------- scalar helpers ----------
__device__ __forceinline__ unsigned bfbits(float f) {   // f32 -> bf16 bits (RNE)
  unsigned u = __float_as_uint(f);
  u += 0x7fffu + ((u >> 16) & 1u);
  return u >> 16;
}
__device__ __forceinline__ __bf16 f2bf(float f) {
  union { unsigned short s; __bf16 b; } cv;
  cv.s = (unsigned short)bfbits(f);
  return cv.b;
}
__device__ __forceinline__ float bf2f(__bf16 b) {
  union { unsigned short s; __bf16 b; } cv;
  cv.b = b;
  return __uint_as_float(((unsigned)cv.s) << 16);
}
__device__ __forceinline__ float siluf(float x) { return x / (1.f + __expf(-x)); }

// ---------- WMMA fragment helpers ----------
// 16x32 bf16 fragment per CDNA5 ISA layout:
// lane L: row = base + (L&15); elems [0..7]  = k0 + (L>>4)*8 + 0..7
//                              elems [8..15] = same + 16
__device__ __forceinline__ v16bf frag_ld(const __bf16* base, int ld, int row, int k0) {
  int lane = threadIdx.x & 31;
  int m  = row + (lane & 15);
  int ko = k0 + ((lane >> 4) << 3);
  const __bf16* q = base + (size_t)m * ld + ko;
  v16bf f;
  uint4* pf = reinterpret_cast<uint4*>(&f);
  pf[0] = *reinterpret_cast<const uint4*>(q);
  pf[1] = *reinterpret_cast<const uint4*>(q + 16);
  return f;
}
__device__ __forceinline__ v8f wmma_bf16(v16bf a, v16bf b, v8f c) {
  return __builtin_amdgcn_wmma_f32_16x16x32_bf16(false, a, false, b, (short)0, c, false, false);
}

// ---------- fp32 -> bf16 convert with zero tail padding ----------
__global__ void k_f2bf(const float* __restrict__ in, __bf16* __restrict__ out,
                       int nsrc, int ndst) {
  int i = blockIdx.x * 256 + threadIdx.x;
  if (i < ndst) out[i] = (i < nsrc) ? f2bf(in[i]) : f2bf(0.f);
}

// ---------- RMSNorm (fp32 in, bf16 out), one token per block ----------
__global__ __launch_bounds__(256) void k_rmsnorm(const float* __restrict__ x,
                                                 const float* __restrict__ w,
                                                 __bf16* __restrict__ out) {
  int tok = blockIdx.x;
  const float* xr = x + (size_t)tok * DMODEL;
  __shared__ float red[256];
  float v[4]; float ss = 0.f;
  for (int i = 0; i < 4; ++i) { v[i] = xr[threadIdx.x + i*256]; ss += v[i]*v[i]; }
  red[threadIdx.x] = ss; __syncthreads();
  for (int o = 128; o > 0; o >>= 1) {
    if (threadIdx.x < o) red[threadIdx.x] += red[threadIdx.x + o];
    __syncthreads();
  }
  float sc = rsqrtf(red[0] / (float)DMODEL + EPS);
  for (int i = 0; i < 4; ++i) {
    int idx = threadIdx.x + i*256;
    out[(size_t)tok*DMODEL + idx] = f2bf(v[i] * sc * w[idx]);
  }
}

// ---------- Generic GEMM: C[M,N] = A[M,K] * W[N,K]^T (+ residual) ----------
// Tile-exact launches only: block tile 64(M) x 256(N), 8 waves = 2(M) x 4(N),
// wave tile 32x64, double-buffered LDS, 1 barrier per K-step.
template<bool HASRES>
__global__ __launch_bounds__(256) void k_gemm_bf16(const __bf16* __restrict__ A,
                                                   const __bf16* __restrict__ W,
                                                   float* __restrict__ C,
                                                   const float* __restrict__ R,
                                                   int M, int N, int K,
                                                   long sA, long sW, long sC) {
  int z = blockIdx.z;
  A += (size_t)z * sA; W += (size_t)z * sW; C += (size_t)z * sC;
  if (HASRES) R += (size_t)z * sC;

  __shared__ __bf16 As[2][64*32];    // 2 x 4 KB
  __shared__ __bf16 Ws[2][256*32];   // 2 x 16 KB
  int tid = threadIdx.x, lane = tid & 31, wid = tid >> 5;
  int wm = wid >> 2, wn = wid & 3;
  int gm0 = blockIdx.y * 64, gn0 = blockIdx.x * 256;

  // per-thread load coords: 4 threads per row, 16 B each (coalesced 64 B / 4 lanes)
  int lrow = tid >> 2, lch = (tid & 3) * 8;
  const __bf16* aptr = A + (size_t)(gm0 + lrow) * K + lch;
  const __bf16* wptr = W + (size_t)(gn0 + lrow) * K + lch;

  uint4 ar; uint4 wr[4];
  ar = *(const uint4*)(aptr);
  for (int g = 0; g < 4; ++g) wr[g] = *(const uint4*)(wptr + (size_t)g*64*K);
  *(uint4*)&As[0][lrow*32 + lch] = ar;
  for (int g = 0; g < 4; ++g) *(uint4*)&Ws[0][(lrow + g*64)*32 + lch] = wr[g];
  __syncthreads();

  v8f acc[2][4] = {};
  int nk = K >> 5;
  for (int ki = 0; ki < nk; ++ki) {
    int cur = ki & 1, nxt = cur ^ 1;
    if (ki + 2 < nk) {   // warm GL2 for the tile after next
      __builtin_prefetch((const void*)(aptr + (ki + 2) * 32), 0, 1);
      __builtin_prefetch((const void*)(wptr + (ki + 2) * 32), 0, 1);
    }
    if (ki + 1 < nk) {   // fetch next tile into registers
      const __bf16* ap = aptr + (ki + 1) * 32;
      const __bf16* wp = wptr + (ki + 1) * 32;
      ar = *(const uint4*)ap;
      for (int g = 0; g < 4; ++g) wr[g] = *(const uint4*)(wp + (size_t)g*64*K);
    }
    v16bf af[2], bfr[4];
    for (int mi = 0; mi < 2; ++mi) af[mi]  = frag_ld(&As[cur][0], 32, wm*32 + mi*16, 0);
    for (int ni = 0; ni < 4; ++ni) bfr[ni] = frag_ld(&Ws[cur][0], 32, wn*64 + ni*16, 0);
    for (int mi = 0; mi < 2; ++mi)
      for (int ni = 0; ni < 4; ++ni)
        acc[mi][ni] = wmma_bf16(af[mi], bfr[ni], acc[mi][ni]);
    if (ki + 1 < nk) {   // commit next tile to the other buffer; single barrier
      *(uint4*)&As[nxt][lrow*32 + lch] = ar;
      for (int g = 0; g < 4; ++g) *(uint4*)&Ws[nxt][(lrow + g*64)*32 + lch] = wr[g];
      __syncthreads();
    }
  }

  int hi = (lane >> 4) << 3, nn = lane & 15;
  for (int mi = 0; mi < 2; ++mi)
    for (int ni = 0; ni < 4; ++ni)
      for (int r = 0; r < 8; ++r) {
        int row = gm0 + wm*32 + mi*16 + r + hi;
        int col = gn0 + wn*64 + ni*16 + nn;
        float v = acc[mi][ni][r];
        if (HASRES) v += R[(size_t)row * N + col];
        C[(size_t)row * N + col] = v;
      }
}

// ---------- depthwise causal conv + SiLU + dt softplus + split ----------
__global__ __launch_bounds__(256) void k_conv(const float* __restrict__ zx,
                                              const float* __restrict__ cw,
                                              const float* __restrict__ cb,
                                              const float* __restrict__ dtb,
                                              const float* __restrict__ alog,
                                              __bf16* __restrict__ xin,
                                              __bf16* __restrict__ xdt,
                                              __bf16* __restrict__ Bo,
                                              __bf16* __restrict__ Co,
                                              float* __restrict__ dtA) {
  int tok = blockIdx.x;
  int b = tok / SEQ, l = tok % SEQ;
  __shared__ float sdt[NHEADS];
  if (threadIdx.x < NHEADS) {
    int h = threadIdx.x;
    float raw = zx[(size_t)tok*NPAD + DINNER + CONVDIM + h] + dtb[h];
    float dt  = raw > 20.f ? raw : log1pf(__expf(raw));
    sdt[h] = dt;
    dtA[(size_t)tok*NHEADS + h] = -dt * __expf(alog[h]);
  }
  __syncthreads();
  for (int ch = threadIdx.x; ch < CONVDIM; ch += 256) {
    float acc = cb[ch];
    for (int k = 0; k < DCONV; ++k) {
      int li = l - (DCONV - 1) + k;
      if (li >= 0)
        acc += zx[(size_t)(b*SEQ + li)*NPAD + DINNER + ch] * cw[ch*DCONV + k];
    }
    float v = siluf(acc);
    if (ch < DINNER) {
      int h = ch >> 6;
      xin[(size_t)tok*DINNER + ch] = f2bf(v);
      xdt[(size_t)tok*DINNER + ch] = f2bf(v * sdt[h]);
    } else if (ch < DINNER + DSTATE) {
      Bo[(size_t)tok*DSTATE + (ch - DINNER)] = f2bf(v);
    } else {
      Co[(size_t)tok*DSTATE + (ch - DINNER - DSTATE)] = f2bf(v);
    }
  }
}

// ---------- per (b,c,h): Y_diag = (G o L) @ Xdt, S_local = Xdt^T (B * decay) ----------
// dynamic LDS: acs[256] | edec[256] | Xt[64][256] bf16 | Wr: 2x Ws[256][32] ping-pong
// (phase 2 reuses the 32 KB Wr region as Btd[64][256])
__global__ __launch_bounds__(256) void k_ydiag(const float* __restrict__ G,
                                               const __bf16* __restrict__ xdt,
                                               const __bf16* __restrict__ Bb,
                                               const float* __restrict__ dtA,
                                               float* __restrict__ Y,
                                               float* __restrict__ Sloc,
                                               float* __restrict__ Aend) {
  int h = blockIdx.x, c = blockIdx.y, b = blockIdx.z;
  int tid = threadIdx.x, lane = tid & 31, w = tid >> 5;
  extern __shared__ char smem[];
  float*  acs  = (float*)smem;                                 // 1 KB
  float*  edec = (float*)(smem + 1024);                        // 1 KB
  __bf16* Xt   = (__bf16*)(smem + 2048);                       // 32 KB, ld=CHUNK
  __bf16* Wr   = (__bf16*)(smem + 2048 + DHEAD*CHUNK*2);       // 32 KB dual-use
  int tok0 = b*SEQ + c*CHUNK;

  // inclusive cumsum of dt*A over the chunk
  acs[tid] = dtA[(size_t)(tok0 + tid)*NHEADS + h];
  __syncthreads();
  for (int off = 1; off < CHUNK; off <<= 1) {
    float v = acs[tid];
    if (tid >= off) v += acs[tid - off];
    __syncthreads(); acs[tid] = v; __syncthreads();
  }
  float aend = acs[CHUNK - 1];
  edec[tid] = __expf(aend - acs[tid]);   // chunk-end decay, one exp per thread

  // Xdt transposed into LDS: Xt[p][s]; b128 global loads, b16 scattered LDS stores
  for (int idx8 = tid; idx8 < CHUNK*8; idx8 += 256) {
    int s = idx8 >> 3, p0 = (idx8 & 7) * 8;
    uint4 raw = *(const uint4*)(xdt + (size_t)(tok0 + s)*DINNER + h*DHEAD + p0);
    const __bf16* xp = (const __bf16*)&raw;
    for (int q = 0; q < 8; ++q) Xt[(p0 + q)*CHUNK + s] = xp[q];
  }
  // (covered by the first barrier inside the phase-1 loop)

  // phase 1: Y_diag. wave w -> rows [w*32, w*32+32), N = 64; ping-pong decay tile
  const float* Gp = G + (size_t)(b*NCH + c) * CHUNK * CHUNK;
  v8f acc[2][4] = {};
  for (int ki = 0; ki < CHUNK/32; ++ki) {
    int s0 = ki * 32;
    __bf16* Wcur = Wr + (ki & 1) * (CHUNK*32);
    { // masked decay tile [256][32]: fully branch-free via sign-extended bit masks
      int r = tid;
      float arow = acs[r];
      const float* grow = Gp + (size_t)r * CHUNK + s0;
      int jlim = r - s0;                 // j <= jlim is causal (unmasked)
      for (int j = 0; j < 32; j += 4) {
        float4 g  = *(const float4*)(grow + j);
        float4 a4 = *(const float4*)(&acs[s0 + j]);
        // keep-mask: all-ones iff (jlim - j - q) >= 0; arithmetic, no selects
        unsigned m0 = ~(unsigned)((jlim - j - 0) >> 31);
        unsigned m1 = ~(unsigned)((jlim - j - 1) >> 31);
        unsigned m2 = ~(unsigned)((jlim - j - 2) >> 31);
        unsigned m3 = ~(unsigned)((jlim - j - 3) >> 31);
        float v0 = __uint_as_float(__float_as_uint(g.x * __expf(arow - a4.x)) & m0);
        float v1 = __uint_as_float(__float_as_uint(g.y * __expf(arow - a4.y)) & m1);
        float v2 = __uint_as_float(__float_as_uint(g.z * __expf(arow - a4.z)) & m2);
        float v3 = __uint_as_float(__float_as_uint(g.w * __expf(arow - a4.w)) & m3);
        uint2 pk;
        pk.x = (bfbits(v1) << 16) | (bfbits(v0) & 0xffffu);
        pk.y = (bfbits(v3) << 16) | (bfbits(v2) & 0xffffu);
        *(uint2*)&Wcur[r*32 + j] = pk;
      }
    }
    __syncthreads();
    v16bf bfr[4];
    for (int ni = 0; ni < 4; ++ni) bfr[ni] = frag_ld(Xt, CHUNK, ni*16, s0);
    for (int mi = 0; mi < 2; ++mi) {
      v16bf af = frag_ld(Wcur, 32, w*32 + mi*16, 0);
      for (int ni = 0; ni < 4; ++ni)
        acc[mi][ni] = wmma_bf16(af, bfr[ni], acc[mi][ni]);
    }
    // no trailing barrier: next iteration writes the other buffer
  }
  int hi = (lane >> 4) << 3, nn = lane & 15;
  for (int mi = 0; mi < 2; ++mi)
    for (int ni = 0; ni < 4; ++ni)
      for (int r = 0; r < 8; ++r) {
        int lrow = w*32 + mi*16 + r + hi;
        int p    = ni*16 + nn;
        Y[(size_t)(tok0 + lrow)*DINNER + h*DHEAD + p] = acc[mi][ni][r];
      }
  __syncthreads();

  // phase 2: S[p][n] = sum_s Xt[p][s] * (B[s][n] * edec[s])
  for (int idx8 = tid; idx8 < CHUNK*8; idx8 += 256) {
    int s = idx8 >> 3, n0 = (idx8 & 7) * 8;
    float dec = edec[s];
    uint4 raw = *(const uint4*)(Bb + (size_t)(tok0 + s)*DSTATE + n0);
    const __bf16* bp = (const __bf16*)&raw;
    for (int q = 0; q < 8; ++q)
      Wr[(n0 + q)*CHUNK + s] = f2bf(bf2f(bp[q]) * dec);
  }
  __syncthreads();
  v8f sacc[2] = {};
  int t0 = w * 2;
  for (int tt = 0; tt < 2; ++tt) {
    int pi = (t0 + tt) >> 2, nj = (t0 + tt) & 3;
    for (int s0 = 0; s0 < CHUNK; s0 += 32) {
      v16bf af  = frag_ld(Xt, CHUNK, pi*16, s0);
      v16bf bfr = frag_ld(Wr, CHUNK, nj*16, s0);
      sacc[tt] = wmma_bf16(af, bfr, sacc[tt]);
    }
  }
  float* Sp = Sloc + ((size_t)(b*NCH + c)*NHEADS + h) * (DHEAD*DSTATE);
  for (int tt = 0; tt < 2; ++tt) {
    int pi = (t0 + tt) >> 2, nj = (t0 + tt) & 3;
    for (int r = 0; r < 8; ++r) {
      int p = pi*16 + r + hi, n = nj*16 + nn;
      Sp[p*DSTATE + n] = sacc[tt][r];
    }
  }
  if (tid == 0) Aend[(size_t)(b*NCH + c)*NHEADS + h] = aend;
}

// ---------- inter-chunk state recurrence (sequential over 8 chunks) ----------
__global__ __launch_bounds__(256) void k_scan(const float* __restrict__ Sloc,
                                              const float* __restrict__ Aend,
                                              float* __restrict__ Sprev) {
  int h = blockIdx.x, b = blockIdx.y;
  int tid = threadIdx.x;
  float carry[16];
  for (int j = 0; j < 16; ++j) carry[j] = 0.f;
  for (int c = 0; c < NCH; ++c) {
    size_t off = ((size_t)(b*NCH + c)*NHEADS + h) * (DHEAD*DSTATE);
    float e = __expf(Aend[(size_t)(b*NCH + c)*NHEADS + h]);
    for (int j = 0; j < 16; ++j) {
      int idx = tid + j*256;
      Sprev[off + idx] = carry[j];
      carry[j] = carry[j]*e + Sloc[off + idx];
    }
  }
}

// ---------- Y += exp(Acs) * C @ states_prev^T + D * X ----------
__global__ __launch_bounds__(256) void k_yoff(const __bf16* __restrict__ Cb,
                                              const float* __restrict__ Sprev,
                                              const float* __restrict__ dtA,
                                              const __bf16* __restrict__ xin,
                                              const float* __restrict__ Dp,
                                              float* __restrict__ Y) {
  int h = blockIdx.x, c = blockIdx.y, b = blockIdx.z;
  int tid = threadIdx.x, lane = tid & 31, w = tid >> 5;
  __shared__ float  acs[CHUNK];
  __shared__ __bf16 spb[DHEAD*DSTATE];
  int tok0 = b*SEQ + c*CHUNK;
  acs[tid] = dtA[(size_t)(tok0 + tid)*NHEADS + h];
  __syncthreads();
  for (int off = 1; off < CHUNK; off <<= 1) {
    float v = acs[tid];
    if (tid >= off) v += acs[tid - off];
    __syncthreads(); acs[tid] = v; __syncthreads();
  }
  size_t soff = ((size_t)(b*NCH + c)*NHEADS + h) * (DHEAD*DSTATE);
  for (int idx = tid; idx < DHEAD*DSTATE; idx += 256)
    spb[idx] = f2bf(Sprev[soff + idx]);
  __syncthreads();

  const __bf16* Cp = Cb + (size_t)tok0 * DSTATE;
  v8f acc[2][4] = {};
  for (int k0 = 0; k0 < DSTATE; k0 += 32) {
    for (int mi = 0; mi < 2; ++mi) {
      v16bf af = frag_ld(Cp, DSTATE, w*32 + mi*16, k0);
      for (int ni = 0; ni < 4; ++ni) {
        v16bf bfr = frag_ld(spb, DSTATE, ni*16, k0);
        acc[mi][ni] = wmma_bf16(af, bfr, acc[mi][ni]);
      }
    }
  }
  float Dh = Dp[h];
  int hi = (lane >> 4) << 3, nn = lane & 15;
  for (int mi = 0; mi < 2; ++mi)
    for (int ni = 0; ni < 4; ++ni)
      for (int r = 0; r < 8; ++r) {
        int lrow = w*32 + mi*16 + r + hi;
        int p    = ni*16 + nn;
        size_t yi = (size_t)(tok0 + lrow)*DINNER + h*DHEAD + p;
        Y[yi] += __expf(acs[lrow]) * acc[mi][ni][r] + Dh * bf2f(xin[yi]);
      }
}

// ---------- gated RMSNorm: yg = rmsnorm(Y * silu(z)) * gw, bf16 out ----------
__global__ __launch_bounds__(256) void k_gate(const float* __restrict__ Y,
                                              const float* __restrict__ zx,
                                              const float* __restrict__ gw,
                                              __bf16* __restrict__ yg) {
  int tok = blockIdx.x;
  __shared__ float red[256];
  float v[8]; float ss = 0.f;
  for (int i = 0; i < 8; ++i) {
    int idx = threadIdx.x + i*256;
    float z = zx[(size_t)tok*NPAD + idx];
    float t = Y[(size_t)tok*DINNER + idx] * siluf(z);
    v[i] = t; ss += t*t;
  }
  red[threadIdx.x] = ss; __syncthreads();
  for (int o = 128; o > 0; o >>= 1) {
    if (threadIdx.x < o) red[threadIdx.x] += red[threadIdx.x + o];
    __syncthreads();
  }
  float sc = rsqrtf(red[0] / (float)DINNER + EPS);
  for (int i = 0; i < 8; ++i) {
    int idx = threadIdx.x + i*256;
    yg[(size_t)tok*DINNER + idx] = f2bf(v[i] * sc * gw[idx]);
  }
}

// ---------- host orchestration ----------
extern "C" void kernel_launch(void* const* d_in, const int* in_sizes, int n_in,
                              void* d_out, int out_size, void* d_ws, size_t ws_size,
                              hipStream_t stream) {
  (void)in_sizes; (void)n_in; (void)out_size; (void)ws_size;
  const float* x0   = (const float*)d_in[0];
  const float* inw  = (const float*)d_in[1];
  const float* cw   = (const float*)d_in[2];
  const float* cb   = (const float*)d_in[3];
  const float* dtb  = (const float*)d_in[4];
  const float* alog = (const float*)d_in[5];
  const float* Dp   = (const float*)d_in[6];
  const float* gw   = (const float*)d_in[7];
  const float* outw = (const float*)d_in[8];
  const float* rmsw = (const float*)d_in[9];

  char* p = (char*)d_ws;
  auto alloc = [&](size_t bytes) -> char* {
    char* r = p; p += (bytes + 255) & ~(size_t)255; return r;
  };
  float*  xb0   = (float*) alloc((size_t)NTOK*DMODEL*4);
  float*  xb1   = (float*) alloc((size_t)NTOK*DMODEL*4);
  __bf16* xn    = (__bf16*)alloc((size_t)NTOK*DMODEL*2);
  __bf16* w1    = (__bf16*)alloc((size_t)NPAD*DMODEL*2);     // zero-padded rows
  __bf16* w2    = (__bf16*)alloc((size_t)DMODEL*DINNER*2);
  float*  zx    = (float*) alloc((size_t)NTOK*NPAD*4);       // padded ldc
  __bf16* xinb  = (__bf16*)alloc((size_t)NTOK*DINNER*2);
  __bf16* xdtb  = (__bf16*)alloc((size_t)NTOK*DINNER*2);
  __bf16* Bb    = (__bf16*)alloc((size_t)NTOK*DSTATE*2);
  __bf16* Cbm   = (__bf16*)alloc((size_t)NTOK*DSTATE*2);
  float*  dtA   = (float*) alloc((size_t)NTOK*NHEADS*4);
  float*  Aend  = (float*) alloc((size_t)BATCH*NCH*NHEADS*4);
  float*  G     = (float*) alloc((size_t)BATCH*NCH*CHUNK*CHUNK*4);
  float*  Sloc  = (float*) alloc((size_t)BATCH*NCH*NHEADS*DHEAD*DSTATE*4);
  float*  Sprev = (float*) alloc((size_t)BATCH*NCH*NHEADS*DHEAD*DSTATE*4);
  float*  Yb    = (float*) alloc((size_t)NTOK*DINNER*4);
  __bf16* yg    = (__bf16*)alloc((size_t)NTOK*DINNER*2);

  const float* xcur = x0;
  for (int layer = 0; layer < 4; ++layer) {
    float* xout = (layer == 3) ? (float*)d_out : ((layer & 1) ? xb1 : xb0);

    k_rmsnorm<<<NTOK, 256, 0, stream>>>(xcur, rmsw + layer*DMODEL, xn);

    int n1s = DINPROJ*DMODEL, n1d = NPAD*DMODEL;
    k_f2bf<<<(n1d + 255)/256, 256, 0, stream>>>(inw + (size_t)layer*n1s, w1, n1s, n1d);
    int n2 = DMODEL*DINNER;
    k_f2bf<<<(n2 + 255)/256, 256, 0, stream>>>(outw + (size_t)layer*n2, w2, n2, n2);

    dim3 g1(NPAD/256, NTOK/64, 1);   // in_proj, padded N, no guards
    k_gemm_bf16<false><<<g1, 256, 0, stream>>>(xn, w1, zx, nullptr,
                                               NTOK, NPAD, DMODEL, 0, 0, 0);

    k_conv<<<NTOK, 256, 0, stream>>>(zx, cw + (size_t)layer*CONVDIM*DCONV,
                                     cb + layer*CONVDIM, dtb + layer*NHEADS,
                                     alog + layer*NHEADS,
                                     xinb, xdtb, Bb, Cbm, dtA);

    dim3 g2(1, CHUNK/64, BATCH*NCH);   // G = C @ B^T per (b,c): 256x256, K=64
    k_gemm_bf16<false><<<g2, 256, 0, stream>>>(Cbm, Bb, G, nullptr,
                                               CHUNK, CHUNK, DSTATE,
                                               (long)CHUNK*DSTATE, (long)CHUNK*DSTATE,
                                               (long)CHUNK*CHUNK);

    size_t smem_ydiag = 2048 + (size_t)DHEAD*CHUNK*2 + (size_t)CHUNK*32*2*2; // 66 KB
    k_ydiag<<<dim3(NHEADS, NCH, BATCH), 256, smem_ydiag, stream>>>(
        G, xdtb, Bb, dtA, Yb, Sloc, Aend);

    k_scan<<<dim3(NHEADS, BATCH), 256, 0, stream>>>(Sloc, Aend, Sprev);

    k_yoff<<<dim3(NHEADS, NCH, BATCH), 256, 0, stream>>>(
        Cbm, Sprev, dtA, xinb, Dp + layer*NHEADS, Yb);

    k_gate<<<NTOK, 256, 0, stream>>>(Yb, zx, gw + layer*DINNER, yg);

    dim3 g3(DMODEL/256, NTOK/64, 1);   // out_proj + residual
    k_gemm_bf16<true><<<g3, 256, 0, stream>>>(yg, w2, xout, xcur,
                                              NTOK, DMODEL, DINNER, 0, 0, 0);
    xcur = xout;
  }
}
